// Caption_26834955665879
// MI455X (gfx1250) — matrix operand order repaired and checked by reading, not compiled, not run on hardware
//
#include <hip/hip_runtime.h>
#include <hip/hip_bf16.h>
#include <math.h>

// ---------------------------------------------------------------------------
// LSTM caption decoder on gfx1250 (MI455X): all GEMMs via v_wmma_f32_16x16x32_f16
// ---------------------------------------------------------------------------

typedef __attribute__((ext_vector_type(16))) _Float16 v16h;
typedef __attribute__((ext_vector_type(8)))  _Float16 v8h;
typedef __attribute__((ext_vector_type(8)))  float    v8f;

#define B_ 128
#define F_ 1536
#define E_ 512
#define H_ 512
#define V_ 10000
#define T_ 40

union V16u { v16h h16; v8h h8[2]; };

// ---------------------------------------------------------------------------
// Pack a row-major fp32 weight W[N,K] (optionally split across two sources on
// K: W0[N,K0] then W1[N,K1]) into f16 WMMA B-fragment tiles, zero-padded up to
// the launched tile count:
//   Bp[((nt*KT + kt)*32 + lane)*16 + i] = (n < Nvalid) ? W[n][k] : 0
//   with n = nt*16 + lane%16, k = kt*32 + 8*(lane/16) + (i&7) + 16*(i>>3)
// (lane%16 indexes N; lane/16 and i index K, AMD table order) so each lane loads
// its whole fragment with one contiguous 32-byte read.
// ---------------------------------------------------------------------------
__global__ void pack_weights(const float* __restrict__ W0, int K0,
                             const float* __restrict__ W1, int K1,
                             _Float16* __restrict__ Bp, int KT, int Nvalid,
                             long total) {
  long idx = (long)blockIdx.x * blockDim.x + threadIdx.x;
  if (idx >= total) return;
  int  i  = (int)(idx & 15);
  int  l  = (int)((idx >> 4) & 31);
  long tt = idx >> 9;
  int  kt = (int)(tt % KT);
  int  nt = (int)(tt / KT);
  int  n  = nt * 16 + (l & 15);
  int  k  = kt * 32 + ((l >> 4) << 3) + (i & 7) + ((i >> 3) << 4);
  float v = 0.0f;
  if (n < Nvalid)
    v = (k < K0) ? W0[(long)n * K0 + k] : W1[(long)n * K1 + (k - K0)];
  Bp[idx] = (_Float16)v;
}

// seq[b][t][e] = f16(emb[seqs[b][t-1]][e]) for t = 1..T-1
__global__ void embed_gather(const int* __restrict__ seqs,
                             const float* __restrict__ emb,
                             _Float16* __restrict__ seq) {
  long idx = (long)blockIdx.x * blockDim.x + threadIdx.x;
  const long total = (long)B_ * (T_ - 1) * E_;
  if (idx >= total) return;
  int  e = (int)(idx & (E_ - 1));
  long r = idx >> 9;                 // E_ == 512
  int  t = (int)(r % (T_ - 1));
  int  b = (int)(r / (T_ - 1));
  int  w = seqs[b * (T_ - 1) + t];
  seq[((long)b * T_ + (t + 1)) * E_ + e] = (_Float16)emb[(long)w * E_ + e];
}

__global__ void bias_sum(const float* __restrict__ a, const float* __restrict__ b,
                         float* __restrict__ o, int n) {
  int i = blockIdx.x * blockDim.x + threadIdx.x;
  if (i < n) o[i] = a[i] + b[i];
}

__device__ __forceinline__ float sigmoidf_(float x) { return 1.0f / (1.0f + expf(-x)); }

// One LSTM cell update. gates = [i | f | g | o] pre-activations, f32 [B,4H].
// h is written as f16 directly into hiddens[b][t][:], which is both the final
// GEMM's A matrix and the next step's recurrent input.
__global__ void lstm_cell(const float* __restrict__ gates, float* __restrict__ c,
                          _Float16* __restrict__ hiddens, int t) {
  int idx = blockIdx.x * blockDim.x + threadIdx.x;
  if (idx >= B_ * H_) return;
  int b = idx >> 9;                  // H_ == 512
  int j = idx & (H_ - 1);
  const float* g = gates + (long)b * 4 * H_;
  float ig = sigmoidf_(g[j]);
  float fg = sigmoidf_(g[H_ + j]);
  float gg = tanhf(g[2 * H_ + j]);
  float og = sigmoidf_(g[3 * H_ + j]);
  float cp = (t == 0) ? 0.0f : c[idx];
  float cn = fg * cp + ig * gg;
  c[idx] = cn;
  float h = og * tanhf(cn);
  hiddens[((long)b * T_ + t) * H_ + j] = (_Float16)h;
}

// ---------------------------------------------------------------------------
// Generic WMMA GEMM:  C[M,N] = A[M, K0+K1] * W^T + bias, W packed as above and
// zero-padded on N so the K-loop runs with NO divergent guards around WMMA.
// A row r = A0[r, 0:K0] ++ A1[r, 0:K1]  (A1 == nullptr -> zeros; A0 may be f32).
// nt0 is forced into an SGPR via readfirstlane so bounds checks are scalar
// branches and B-fragment addresses are SGPR-base + lane*32B. B fragments are
// double-buffered so global loads for kt+1 overlap the kt WMMAs.
// ---------------------------------------------------------------------------
template<int BM, int WAVES, int NT, bool A_F32, bool OUT_F16>
__global__ __launch_bounds__(WAVES * 32)
void gemm_wmma(const void* __restrict__ A0_, int lda0, int K0,
               const _Float16* __restrict__ A1, int lda1, int K1,
               const _Float16* __restrict__ Bp, const float* __restrict__ bias,
               void* __restrict__ C_, int ldc, int NtilesValid) {
  extern __shared__ char smem_raw[];
  _Float16* As = (_Float16*)smem_raw;

  const int Ktot  = K0 + K1;
  const int pitch = Ktot + 8;        // halves; keeps ds_load_b128 lanes on distinct banks
  const int KT    = Ktot >> 5;
  const int tid   = threadIdx.x;
  const int wave  = tid >> 5;
  const int lane  = tid & 31;
  const int m0    = blockIdx.y * BM;
  // wave-scalar first N-tile: keeps all downstream control flow scalar
  const int nt0   = __builtin_amdgcn_readfirstlane(blockIdx.x * (WAVES * NT) + wave * NT);

  // --- stage A strip (with f32->f16 conversion if needed) ---
  const float*    A0f = (const float*)A0_;
  const _Float16* A0h = (const _Float16*)A0_;
  for (int idx = tid; idx < BM * Ktot; idx += WAVES * 32) {
    int r = idx / Ktot;
    int k = idx - r * Ktot;
    float v;
    if (k < K0) {
      long off = (long)(m0 + r) * lda0 + k;
      v = A_F32 ? A0f[off] : (float)A0h[off];
    } else {
      v = A1 ? (float)A1[(long)(m0 + r) * lda1 + (k - K0)] : 0.0f;
    }
    As[r * pitch + k] = (_Float16)v;
  }
  __syncthreads();

  constexpr int MT = BM / 16;
  v8f acc[MT][NT];
#pragma unroll
  for (int nj = 0; nj < NT; ++nj) {
    float bv = 0.0f;
    if (nt0 + nj < NtilesValid)                     // scalar branch
      bv = bias[(nt0 + nj) * 16 + (lane & 15)];
#pragma unroll
    for (int mi = 0; mi < MT; ++mi)
#pragma unroll
      for (int q = 0; q < 8; ++q) acc[mi][nj][q] = bv;
  }

  // B fragments: SGPR base for this wave's tiles, double-buffered loads
  const _Float16* BpW = Bp + (long)nt0 * KT * 512;
  const int laneB = lane << 4;
  V16u bf[2][NT];
#pragma unroll
  for (int nj = 0; nj < NT; ++nj)
    bf[0][nj].h16 = *(const v16h*)(BpW + ((nj * KT + 0) << 9) + laneB);

  for (int kt = 0; kt < KT; ++kt) {
    const int cur = kt & 1;
    if (kt + 1 < KT) {                              // scalar branch
#pragma unroll
      for (int nj = 0; nj < NT; ++nj)
        bf[cur ^ 1][nj].h16 =
            *(const v16h*)(BpW + (((nj * KT) + (kt + 1)) << 9) + laneB);
    }
    // A fragment (ISA 16-bit 16x32 layout): row = lane%16,
    // halves i<8 -> K = kt*32 + (lane/16)*8 + i ; i>=8 -> +16.
    V16u af[MT];
    const int arow = lane & 15;
    const int kb   = (kt << 5) + ((lane >> 4) << 3);
#pragma unroll
    for (int mi = 0; mi < MT; ++mi) {
      const _Float16* p = &As[(mi * 16 + arow) * pitch + kb];
      af[mi].h8[0] = *(const v8h*)(p);
      af[mi].h8[1] = *(const v8h*)(p + 16);
    }
#pragma unroll
    for (int nj = 0; nj < NT; ++nj)
#pragma unroll
      for (int mi = 0; mi < MT; ++mi)
        acc[mi][nj] = __builtin_amdgcn_wmma_f32_16x16x32_f16(
            false, af[mi].h16, false, bf[cur][nj].h16, (short)0, acc[mi][nj],
            false, false);
  }

  // --- store (C/D layout: lane<16 -> M = v, N = lane; lane>=16 -> M = v+8) ---
#pragma unroll
  for (int nj = 0; nj < NT; ++nj) {
    if (nt0 + nj >= NtilesValid) continue;          // scalar branch
    const int n = (nt0 + nj) * 16 + (lane & 15);
#pragma unroll
    for (int mi = 0; mi < MT; ++mi) {
#pragma unroll
      for (int v = 0; v < 8; ++v) {
        int  m   = m0 + mi * 16 + v + ((lane >> 4) << 3);
        long off = (long)m * ldc + n;
        if (OUT_F16) ((_Float16*)C_)[off] = (_Float16)acc[mi][nj][v];
        else         ((float*)C_)[off]    = acc[mi][nj][v];
      }
    }
  }
}

// ---------------------------------------------------------------------------
extern "C" void kernel_launch(void* const* d_in, const int* in_sizes, int n_in,
                              void* d_out, int out_size, void* d_ws, size_t ws_size,
                              hipStream_t stream) {
  (void)in_sizes; (void)n_in; (void)out_size; (void)ws_size;

  const float* features = (const float*)d_in[0];
  const int*   seqs     = (const int*)d_in[1];
  /* d_in[2] lengths: unused in forward */
  const float* W_in  = (const float*)d_in[3];
  const float* b_in  = (const float*)d_in[4];
  const float* emb   = (const float*)d_in[5];
  const float* W_ih  = (const float*)d_in[6];
  const float* W_hh  = (const float*)d_in[7];
  const float* b_ih  = (const float*)d_in[8];
  const float* b_hh  = (const float*)d_in[9];
  const float* W_out = (const float*)d_in[10];
  const float* b_out = (const float*)d_in[11];
  float* out = (float*)d_out;

  const int VT_PAD = 640;            // W_out N-tiles padded to 16-tile blocks

  // workspace carve-up (~29 MB)
  char*  ws  = (char*)d_ws;
  size_t off = 0;
  auto alloc = [&](size_t bytes) -> void* {
    void* p = ws + off;
    off = (off + bytes + 255) & ~(size_t)255;
    return p;
  };
  _Float16* seqbuf = (_Float16*)alloc((size_t)B_ * T_ * E_ * 2);      // f16 [B,T,E]
  _Float16* hid    = (_Float16*)alloc((size_t)B_ * T_ * H_ * 2);      // f16 [B,T,H]
  _Float16* WinP   = (_Float16*)alloc((size_t)E_ * F_ * 2);           // packed W_in
  _Float16* WcP    = (_Float16*)alloc((size_t)4 * H_ * (E_ + H_) * 2);// packed [W_ih|W_hh]
  _Float16* WoP    = (_Float16*)alloc((size_t)VT_PAD * 16 * H_ * 2);  // packed W_out (padded)
  float*    gates  = (float*)alloc((size_t)B_ * 4 * H_ * 4);          // f32 [B,4H]
  float*    cbuf   = (float*)alloc((size_t)B_ * H_ * 4);              // f32 cell state
  float*    bsum   = (float*)alloc((size_t)4 * H_ * 4);               // b_ih + b_hh

  // --- pack weights into WMMA B-fragment layout ---
  {
    long tot = (long)(E_ / 16) * (F_ / 32) * 512;
    pack_weights<<<(int)((tot + 255) / 256), 256, 0, stream>>>(
        W_in, F_, nullptr, 0, WinP, F_ / 32, E_, tot);
  }
  {
    long tot = (long)(4 * H_ / 16) * ((E_ + H_) / 32) * 512;
    pack_weights<<<(int)((tot + 255) / 256), 256, 0, stream>>>(
        W_ih, E_, W_hh, H_, WcP, (E_ + H_) / 32, 4 * H_, tot);
  }
  {
    long tot = (long)VT_PAD * (H_ / 32) * 512;
    pack_weights<<<(int)((tot + 255) / 256), 256, 0, stream>>>(
        W_out, H_, nullptr, 0, WoP, H_ / 32, V_, tot);
  }
  bias_sum<<<(4 * H_ + 255) / 256, 256, 0, stream>>>(b_ih, b_hh, bsum, 4 * H_);

  // --- build input sequence: t=0 is the projected image feature, t>=1 embeddings ---
  {
    long tot = (long)B_ * (T_ - 1) * E_;
    embed_gather<<<(int)((tot + 255) / 256), 256, 0, stream>>>(seqs, emb, seqbuf);
  }
  {
    // x0 = features @ W_in^T + b_in -> seq[:,0,:] (f16, row stride T*E)
    dim3 grid((E_ / 16) / (4 * 2), B_ / 16);              // (4, 8)
    size_t smem = (size_t)16 * (F_ + 8) * sizeof(_Float16);
    gemm_wmma<16, 4, 2, true, true><<<grid, 4 * 32, smem, stream>>>(
        features, F_, F_, nullptr, 0, 0, WinP, b_in, seqbuf, T_ * E_, E_ / 16);
  }

  // --- LSTM scan: gates = [x_t , h_{t-1}] @ [W_ih|W_hh]^T + (b_ih+b_hh) ---
  for (int t = 0; t < T_; ++t) {
    dim3 grid((4 * H_ / 16) / (8 * 2), B_ / 16);          // (8, 8)
    size_t smem = (size_t)16 * (E_ + H_ + 8) * sizeof(_Float16);
    gemm_wmma<16, 8, 2, false, false><<<grid, 8 * 32, smem, stream>>>(
        seqbuf + (long)t * E_, T_ * E_, E_,
        (t == 0) ? nullptr : (hid + (long)(t - 1) * H_), T_ * H_, H_,
        WcP, bsum, gates, 4 * H_, 4 * H_ / 16);
    lstm_cell<<<(B_ * H_ + 255) / 256, 256, 0, stream>>>(gates, cbuf, hid, t);
  }

  // --- logits = hiddens @ W_out^T + b_out : [5120, 10000] f32 ---
  {
    dim3 grid(VT_PAD / (8 * 2), (B_ * T_) / 32);          // (40, 160)
    size_t smem = (size_t)32 * (H_ + 8) * sizeof(_Float16);
    gemm_wmma<32, 8, 2, false, false><<<grid, 8 * 32, smem, stream>>>(
        hid, H_, H_, nullptr, 0, 0, WoP, b_out, out, V_, V_ / 16);
  }
}
